// MultiTaskTwistorLNN_64098091925871
// MI455X (gfx1250) — matrix operand order repaired and checked by reading, not compiled, not run on hardware
//
#include <hip/hip_runtime.h>
#include <cstddef>

// ---- problem constants (match reference) ----
#define TT     1024
#define BB     512
#define D_IN   64
#define D_OUT  32
#define HH     128
#define EE     16
#define LDWX   (HH + EE)   // 144

// ---- WMMA vector types ----
typedef __attribute__((ext_vector_type(16))) __bf16 v16bf;
typedef __attribute__((ext_vector_type(8)))  __bf16 v8bf;
typedef __attribute__((ext_vector_type(8)))  float  v8f;

// padded LDS strides (in bf16 elements); multiples of 8 keep 16B alignment,
// and (2*stride/4) mod 64 != 0 avoids bank conflicts across the 16 rows.
#define AS 136
#define XS 72

union alignas(16) SmemU {
    float win[HH * D_IN];                 // 32KB: W_in staging, init only
    struct {
        __bf16 tr[16 * AS];               // tanh(zr)  A-matrix
        __bf16 ti[16 * AS];               // tanh(zi)  A-matrix
        __bf16 zm[16 * AS];               // |z|       A-matrix
        __bf16 ya[16 * AS];               // zr2       A-matrix (for y GEMM)
        __bf16 xa[16 * XS];               // x tile    A-matrix (K=64)
    } s;
};

static __device__ __forceinline__ __bf16 f2bf(float f) {
    unsigned u = __builtin_bit_cast(unsigned, f);
    u += 0x7FFFu + ((u >> 16) & 1u);      // round to nearest even
    unsigned short h = (unsigned short)(u >> 16);
    return __builtin_bit_cast(__bf16, h);
}

static __device__ __forceinline__ float sigmoidf_(float x) {
    return 1.0f / (1.0f + __expf(-x));
}

// A-fragment load from LDS, 16x32 bf16, contiguous K order:
// lane<16: row=lane, K in {kb..kb+15} with kb=0; lane>=16: kb=16.
static __device__ __forceinline__ v16bf load_a(const __bf16* base, int stride,
                                               int kc, int lane) {
    int r  = lane & 15;
    int kb = (lane >> 4) << 4;            // 0 or 16
    const __bf16* p = base + r * stride + kc * 32 + kb;
    v8bf lo = *(const v8bf*)p;
    v8bf hi = *(const v8bf*)(p + 8);
    return __builtin_shufflevector(lo, hi, 0, 1, 2, 3, 4, 5, 6, 7,
                                           8, 9, 10, 11, 12, 13, 14, 15);
}

// B-fragment (32x16 bf16): lane<16 -> K=0..15, lane>=16 -> K=16..31; n=lane&15.
// We need B[k][n] = W[row=n_global][k], i.e. read row n of W contiguously in k.
static __device__ __forceinline__ v16bf make_bfrag(const float* W, int row,
                                                   int ld, int kc, int kb16) {
    const float* p = W + (size_t)row * ld + kc * 32 + kb16;
    v16bf r;
#pragma unroll
    for (int i = 0; i < 16; ++i) r[i] = f2bf(p[i]);
    return r;
}

static __device__ __forceinline__ v8f wmma_bf(v16bf a, v16bf b, v8f c) {
    return __builtin_amdgcn_wmma_f32_16x16x32_bf16(
        false, a, false, b, (short)0, c, false, false);
}

__global__ __launch_bounds__(256)
void MultiTaskTwistorLNN_kernel(
    const float* __restrict__ x,        const float* __restrict__ emb,
    const float* __restrict__ Win,      const float* __restrict__ bin,
    const float* __restrict__ Wz,       const float* __restrict__ bz_g,
    const float* __restrict__ Wx,       const float* __restrict__ bx_g,
    const float* __restrict__ Wtau,     const float* __restrict__ btau_g,
    const float* __restrict__ taub_g,   const float* __restrict__ Wg,
    const float* __restrict__ bg_g,     const float* __restrict__ Wout,
    const float* __restrict__ bout_g,   float* __restrict__ out)
{
    __shared__ SmemU lds;

    const int tid  = threadIdx.x;
    const int wv   = tid >> 5;            // wave 0..7 -> output-column tile
    const int lane = tid & 31;
    const int nl   = lane & 15;
    const int ng   = wv * 16 + nl;        // global H column owned by this lane
    const int kb16 = (lane < 16) ? 0 : 16;
    const int b0   = blockIdx.x * 16;     // batch tile base

    // ---------------- init: stage W_in into LDS ----------------
    for (int e = tid; e < HH * D_IN; e += 256) lds.win[e] = Win[e];
    __syncthreads();

    // ---- fuse W_eff = W_x[:, :H] @ W_in, build B-fragments in VGPRs ----
    float weacc[2][16];
#pragma unroll
    for (int kc = 0; kc < 2; ++kc)
#pragma unroll
        for (int i = 0; i < 16; ++i) weacc[kc][i] = 0.0f;
    for (int k = 0; k < HH; ++k) {
        float wx = Wx[(size_t)ng * LDWX + k];
        const float* wr_ = &lds.win[k * D_IN];
#pragma unroll
        for (int kc = 0; kc < 2; ++kc) {
            const float* p = wr_ + kc * 32 + kb16;
#pragma unroll
            for (int i = 0; i < 16; ++i) weacc[kc][i] += wx * p[i];
        }
    }
    v16bf weff[2];
#pragma unroll
    for (int kc = 0; kc < 2; ++kc) {
        v16bf f;
#pragma unroll
        for (int i = 0; i < 16; ++i) f[i] = f2bf(weacc[kc][i]);
        weff[kc] = f;
    }

    // ---- recurrent weight B-fragments (register resident) ----
    v16bf wzf[4], wtf[4];
#pragma unroll
    for (int kc = 0; kc < 4; ++kc) {
        wzf[kc] = make_bfrag(Wz,   ng, HH, kc, kb16);
        wtf[kc] = make_bfrag(Wtau, ng, HH, kc, kb16);
    }
    v16bf wof[4];
    float bo = 0.0f;
    if (wv < 2) {
        int og = wv * 16 + nl;
#pragma unroll
        for (int kc = 0; kc < 4; ++kc)
            wof[kc] = make_bfrag(Wout, og, HH, kc, kb16);
        bo = bout_g[og];
    }

    // ---- per-lane scalar constants for column ng ----
    float uc = bx_g[ng];                  // b_x + W_xl@b_in + W_xe@emb
    for (int j = 0; j < LDWX; ++j)
        uc += Wx[(size_t)ng * LDWX + j] * (j < HH ? bin[j] : emb[j - HH]);
    float gacc = bg_g[ng];
#pragma unroll
    for (int e = 0; e < EE; ++e) gacc += Wg[ng * EE + e] * emb[e];
    const float gate = sigmoidf_(gacc);
    const float bz = bz_g[ng], bt = btau_g[ng], tb = taub_g[ng];

    __syncthreads();                      // LDS union is reused from here on

    // ---- state fragments: zr, zi start at 0 ----
    v8f zr = {}, zi = {};

    // ---------------- time scan ----------------
#pragma unroll 1
    for (int t = 0; t < TT; ++t) {
        // x tile -> bf16 A-matrix (16 x 64); one float4 per thread, coalesced
        {
            int m  = tid >> 4;
            int d4 = (tid & 15) << 2;
            const float* xp = x + ((size_t)t * BB + b0 + m) * D_IN + d4;
            float4 v = *(const float4*)xp;
            __bf16* dst = &lds.s.xa[m * XS + d4];
            dst[0] = f2bf(v.x); dst[1] = f2bf(v.y);
            dst[2] = f2bf(v.z); dst[3] = f2bf(v.w);
            if (t + 1 < TT)
                __builtin_prefetch(xp + (size_t)BB * D_IN, 0, 0); // global_prefetch
        }
        // state-derived A-matrices
#pragma unroll
        for (int v = 0; v < 8; ++v) {
            int m = (lane < 16) ? v : v + 8;
            float r_ = zr[v], i_ = zi[v];
            lds.s.tr[m * AS + ng] = f2bf(tanhf(r_));
            lds.s.ti[m * AS + ng] = f2bf(tanhf(i_));
            float m2 = r_ * r_ + i_ * i_;
            lds.s.zm[m * AS + ng] = f2bf(m2 > 0.0f ? sqrtf(m2) : 0.0f);
        }
        __syncthreads();

        // GEMMs: Ux (K=64), wr/wi (K=128, shared W_z), tau (K=128)
        v8f aux = {}, awr = {}, awi = {}, atu = {};
#pragma unroll
        for (int kc = 0; kc < 2; ++kc)
            aux = wmma_bf(load_a(lds.s.xa, XS, kc, lane), weff[kc], aux);
#pragma unroll
        for (int kc = 0; kc < 4; ++kc) {
            awr = wmma_bf(load_a(lds.s.tr, AS, kc, lane), wzf[kc], awr);
            awi = wmma_bf(load_a(lds.s.ti, AS, kc, lane), wzf[kc], awi);
            atu = wmma_bf(load_a(lds.s.zm, AS, kc, lane), wtf[kc], atu);
        }

        // elementwise ODE update + stage zr2 for the output GEMM
#pragma unroll
        for (int v = 0; v < 8; ++v) {
            float r_ = zr[v], i_ = zi[v];
            float ux  = aux[v] + uc;
            float wrv = awr[v] + bz;
            float wiv = awi[v] + bz;
            float tau = fminf(fmaxf(sigmoidf_(atu[v] + bt) + tb, 0.01f), 1.0f)
                        + 1e-6f;
            float dzr = fminf(fmaxf((-r_ + gate * (wrv + ux)) / tau, -10.f), 10.f);
            float dzi = fminf(fmaxf((-i_ + gate * (wiv + ux)) / tau, -10.f), 10.f);
            r_ = fminf(fmaxf(r_ + 0.1f * dzr, -100.f), 100.f);
            i_ = fminf(fmaxf(i_ + 0.1f * dzi, -100.f), 100.f);
            zr[v] = r_; zi[v] = i_;
            int m = (lane < 16) ? v : v + 8;
            lds.s.ya[m * AS + ng] = f2bf(r_);
        }
        __syncthreads();

        // y_t = zr2 @ W_out.T + b_out  (N=32 -> waves 0..1, wave-uniform branch)
        if (wv < 2) {
            v8f ay = {};
#pragma unroll
            for (int kc = 0; kc < 4; ++kc)
                ay = wmma_bf(load_a(lds.s.ya, AS, kc, lane), wof[kc], ay);
#pragma unroll
            for (int v = 0; v < 8; ++v) {
                int m = (lane < 16) ? v : v + 8;
                out[((size_t)t * BB + b0 + m) * D_OUT + wv * 16 + nl]
                    = ay[v] + bo;
            }
        }
    }
}

extern "C" void kernel_launch(void* const* d_in, const int* in_sizes, int n_in,
                              void* d_out, int out_size, void* d_ws, size_t ws_size,
                              hipStream_t stream) {
    (void)in_sizes; (void)n_in; (void)out_size; (void)d_ws; (void)ws_size;
    const float* x      = (const float*)d_in[0];
    const float* emb    = (const float*)d_in[1];
    const float* Win    = (const float*)d_in[2];
    const float* bin    = (const float*)d_in[3];
    const float* Wz     = (const float*)d_in[4];
    const float* bz     = (const float*)d_in[5];
    const float* Wx     = (const float*)d_in[6];
    const float* bx     = (const float*)d_in[7];
    const float* Wtau   = (const float*)d_in[8];
    const float* btau   = (const float*)d_in[9];
    const float* taub   = (const float*)d_in[10];
    const float* Wg     = (const float*)d_in[11];
    const float* bg     = (const float*)d_in[12];
    const float* Wout   = (const float*)d_in[13];
    const float* bout   = (const float*)d_in[14];
    float* out          = (float*)d_out;

    dim3 grid(BB / 16), block(256);
    hipLaunchKernelGGL(MultiTaskTwistorLNN_kernel, grid, block, 0, stream,
                       x, emb, Win, bin, Wz, bz, Wx, bx, Wtau, btau,
                       taub, Wg, bg, Wout, bout, out);
}
